// QLSTM_65481071403379
// MI455X (gfx1250) — matrix operand order repaired and checked by reading, not compiled, not run on hardware
//
#include <hip/hip_runtime.h>
#include <hip/hip_bf16.h>

// LSTM on MI455X (gfx1250):
//  phase 0a: W (4x[512,1024] f32) -> bf16, layout [gate][n][k] (k contiguous)
//  phase 0b: x (f32 [32768,512]) -> bf16 (bandwidth-bound, keeps cvt VALU out of GEMM)
//  phase 1 : z = x @ Wx^T for all T in parallel; each wave owns a 64x32 C block
//            (4 A-frags x 2 B-frags -> 8 WMMAs per 32-wide K chunk)
//  phase 2 : 16 persistent WGs, disjoint 32-column weight slices (L2-resident);
//            h_{t-1} staged through LDS in two 33KB half-K stages (ds_load A-frags),
//            c in registers in C-fragment layout, grid barrier per step.

#define T_STEPS 512
#define BATCH   64
#define DIN     512
#define DH      512
#define LDW     (DIN + DH)      // 1024, k-stride of converted weights
#define NWG2    16              // workgroups in recurrent kernel
#define COLS_WG (DH / NWG2)     // 32 hidden columns per WG
#define HK      256             // K half staged in LDS per phase
#define LDS_LDK (HK + 8)        // 264 u16: 16 A-lanes land in distinct banks

typedef __attribute__((ext_vector_type(16))) __bf16       v16bf;
typedef __attribute__((ext_vector_type(8)))  float        v8f;
typedef __attribute__((ext_vector_type(8)))  unsigned int v8u;

__device__ __forceinline__ unsigned short f2bf(float x) {
  unsigned u = __builtin_bit_cast(unsigned, x);
  u += 0x7FFFu + ((u >> 16) & 1u);          // round-to-nearest-even
  return (unsigned short)(u >> 16);
}
__device__ __forceinline__ float bf2f(unsigned short s) {
  unsigned u = ((unsigned)s) << 16;
  return __builtin_bit_cast(float, u);
}
__device__ __forceinline__ float sigf(float x) { return 1.0f / (1.0f + __expf(-x)); }
__device__ __forceinline__ float tanh_fast(float x) {
  float e = __expf(2.0f * x);
  return 1.0f - 2.0f / (e + 1.0f);          // saturates correctly at +-1
}
// A-fragment K index for VGPR v, lane-half h (ISA 16x32 bf16 layout)
__device__ __forceinline__ int a_k(int v, int half) {
  return (v < 4) ? (2 * v + 8 * half) : (16 + 2 * (v - 4) + 8 * half);
}

// A-matrix (16x32 bf16) fragment from global bf16, row-major [M][ldk]
__device__ __forceinline__ v16bf load_a_bf16(const unsigned short* src, int row,
                                             int ldk, int kbase, int half) {
  v8u u;
#pragma unroll
  for (int v = 0; v < 8; ++v)
    u[v] = *(const unsigned int*)(src + (size_t)row * ldk + kbase + a_k(v, half));
  return __builtin_bit_cast(v16bf, u);
}

// B-matrix (32x16 bf16) fragment: lane holds N=L%16; VGPR v: K = a_k(v, L/16), +1.
__device__ __forceinline__ v16bf load_b_bf16(const unsigned short* src, int n,
                                             int ldk, int kbase, int half) {
  v8u u;
#pragma unroll
  for (int v = 0; v < 8; ++v)
    u[v] = *(const unsigned int*)(src + (size_t)n * ldk + kbase + a_k(v, half));
  return __builtin_bit_cast(v16bf, u);
}

// ---------------- phase 0a: weight conversion ----------------
__global__ void lstm_convert_w(const float* Wf, const float* Wi, const float* Wg,
                               const float* Wo, unsigned short* wbf) {
  size_t idx = (size_t)blockIdx.x * blockDim.x + threadIdx.x;
  const size_t per = (size_t)DH * LDW;
  if (idx >= 4 * per) return;
  int gate = (int)(idx / per);
  size_t off = idx % per;
  const float* W = (gate == 0) ? Wf : (gate == 1) ? Wi : (gate == 2) ? Wg : Wo;
  wbf[idx] = f2bf(W[off]);
}

// ---------------- phase 0b: x conversion (8 floats / thread) ----------------
__global__ void lstm_convert_x(const float* __restrict__ x,
                               unsigned short* __restrict__ xbf) {
  size_t i = ((size_t)blockIdx.x * blockDim.x + threadIdx.x) * 8;
  float4 f0 = *(const float4*)(x + i);
  float4 f1 = *(const float4*)(x + i + 4);
  uint4 o;
  o.x = (unsigned)f2bf(f0.x) | ((unsigned)f2bf(f0.y) << 16);
  o.y = (unsigned)f2bf(f0.z) | ((unsigned)f2bf(f0.w) << 16);
  o.z = (unsigned)f2bf(f1.x) | ((unsigned)f2bf(f1.y) << 16);
  o.w = (unsigned)f2bf(f1.z) | ((unsigned)f2bf(f1.w) << 16);
  *(uint4*)(xbf + i) = o;
}

// ---------------- phase 1: z = x @ Wx^T (parallel over T) ----------------
// Wave grid per WG: 2 (M) x 4 (N); each wave owns 64x32 of C (4x2 tiles).
// WG block = 128 M x 128 N; N blocks never cross a gate boundary (512 | 128*4).
__global__ __launch_bounds__(256) void lstm_xgemm(const unsigned short* __restrict__ xbf,
                                                  const unsigned short* __restrict__ wbf,
                                                  unsigned short* __restrict__ zbuf) {
  int lane = threadIdx.x & 31, wave = threadIdx.x >> 5;
  int half = lane >> 4, l16 = lane & 15;
  int m0 = blockIdx.y * 128 + (wave & 1) * 64;   // wave M base (4 tiles)
  int n0 = blockIdx.x * 128 + (wave >> 1) * 32;  // wave N base (2 tiles), global col
  int gate = n0 >> 9;
  int nb = n0 & 511;
  const unsigned short* bsrc = wbf + (size_t)gate * DH * LDW;

  v8f acc[4][2] = {};
  for (int kk = 0; kk < DIN / 32; ++kk) {
    int kbase = kk * 32;
    v16bf a[4], b[2];
#pragma unroll
    for (int mi = 0; mi < 4; ++mi)
      a[mi] = load_a_bf16(xbf, m0 + mi * 16 + l16, DIN, kbase, half);
#pragma unroll
    for (int ni = 0; ni < 2; ++ni)
      b[ni] = load_b_bf16(bsrc, nb + ni * 16 + l16, LDW, kbase, half);
#pragma unroll
    for (int mi = 0; mi < 4; ++mi)
#pragma unroll
      for (int ni = 0; ni < 2; ++ni)
        acc[mi][ni] = __builtin_amdgcn_wmma_f32_16x16x32_bf16(
            false, a[mi], false, b[ni], (short)0, acc[mi][ni], false, false);
  }
  // store bf16 into zbuf[T][4][B][512]; per VGPR: M = v + 8*half, N = l16.
#pragma unroll
  for (int mi = 0; mi < 4; ++mi)
#pragma unroll
    for (int ni = 0; ni < 2; ++ni) {
      int col = nb + ni * 16 + l16;
#pragma unroll
      for (int v = 0; v < 8; ++v) {
        int row = m0 + mi * 16 + v + 8 * half;   // tb
        int t = row >> 6, bb = row & 63;
        zbuf[(((size_t)t * 4 + gate) * BATCH + bb) * DH + col] = f2bf(acc[mi][ni][v]);
      }
    }
}

// ---------------- phase 2: recurrent steps (persistent, grid-synced) ----------------
__global__ __launch_bounds__(256) void lstm_recurrent(
    const unsigned short* __restrict__ zbuf, const unsigned short* __restrict__ wbf,
    const float* __restrict__ bfv, const float* __restrict__ biv,
    const float* __restrict__ bgv, const float* __restrict__ bov,
    unsigned short* __restrict__ hbuf, float* __restrict__ out,
    unsigned int* __restrict__ ctr) {
  __shared__ __align__(16) unsigned short hlds[BATCH * LDS_LDK];  // 33 KB half-K stage

  int lane = threadIdx.x & 31, wave = threadIdx.x >> 5;
  int half = lane >> 4, l16 = lane & 15;
  int m0 = (wave >> 1) * 16;                      // 4 batch tiles x 2 col tiles
  int n0 = blockIdx.x * COLS_WG + (wave & 1) * 16;
  int col = n0 + l16;

  float biasv[4] = { bfv[col], biv[col], bgv[col], bov[col] };

  const unsigned short* wsrc[4];
#pragma unroll
  for (int g = 0; g < 4; ++g)
    wsrc[g] = wbf + (size_t)g * DH * LDW + DIN;   // hidden half: k offset +512

  // LDS copy assignment: thread -> (row, quarter of a 256-wide K half)
  int crow = threadIdx.x >> 2, cq = threadIdx.x & 3;

  float cfrag[8];
#pragma unroll
  for (int i = 0; i < 8; ++i) cfrag[i] = 0.0f;

  for (int t = 0; t < T_STEPS; ++t) {
    const unsigned short* hrd = hbuf + (size_t)((t + 1) & 1) * BATCH * DH; // h_{t-1}
    unsigned short*       hwr = hbuf + (size_t)(t & 1) * BATCH * DH;       // h_t

    // accumulators = precomputed x-part + bias (C-fragment layout)
    v8f acc[4];
#pragma unroll
    for (int g = 0; g < 4; ++g) {
      v8f a;
#pragma unroll
      for (int v = 0; v < 8; ++v) {
        int b = m0 + v + 8 * half;
        a[v] = bf2f(zbuf[(((size_t)t * 4 + g) * BATCH + b) * DH + col]) + biasv[g];
      }
      acc[g] = a;
    }

    // acc += h_{t-1} @ Wh^T, K staged through LDS in two 256-wide halves
    for (int ph = 0; ph < 2; ++ph) {
      __syncthreads();                            // WAR guard on hlds
      {
        const unsigned short* src = hrd + (size_t)crow * DH + ph * HK + cq * 64;
#pragma unroll
        for (int i = 0; i < 8; ++i)
          ((uint4*)(hlds + crow * LDS_LDK + cq * 64))[i] = ((const uint4*)src)[i];
      }
      __syncthreads();
#pragma unroll 2
      for (int kk = 0; kk < HK / 32; ++kk) {
        int klocal = kk * 32;
        int kglob = ph * HK + klocal;
        __builtin_prefetch(wsrc[0] + (size_t)col * LDW + kglob + 32, 0, 0);
        // A fragment from LDS (padded stride: conflict-light ds_load_b32)
        v8u au;
#pragma unroll
        for (int v = 0; v < 8; ++v)
          au[v] = *(const unsigned int*)(hlds + (m0 + l16) * LDS_LDK + klocal + a_k(v, half));
        v16bf afrag = __builtin_bit_cast(v16bf, au);
#pragma unroll
        for (int g = 0; g < 4; ++g) {
          v16bf bfrag = load_b_bf16(wsrc[g], col, LDW, kglob, half);
          acc[g] = __builtin_amdgcn_wmma_f32_16x16x32_bf16(
              false, afrag, false, bfrag, (short)0, acc[g], false, false);
        }
      }
    }

    // gate nonlinearities + state update, all in fragment layout
    float hfrag[8];
#pragma unroll
    for (int i = 0; i < 8; ++i) {
      float f  = sigf(acc[0][i]);
      float ig = sigf(acc[1][i]);
      float g  = tanh_fast(acc[2][i]);
      float o  = sigf(acc[3][i]);
      float c  = f * cfrag[i] + ig * g;
      cfrag[i] = c;
      hfrag[i] = o * tanh_fast(c);
    }

    // write h: bf16 ping-pong stage + fp32 output row
#pragma unroll
    for (int v = 0; v < 8; ++v) {
      int b = m0 + v + 8 * half;
      hwr[(size_t)b * DH + col] = f2bf(hfrag[v]);
      out[((size_t)t * BATCH + b) * DH + col] = hfrag[v];
    }
    if (t == T_STEPS - 1) {
      size_t base = (size_t)T_STEPS * BATCH * DH;
#pragma unroll
      for (int v = 0; v < 8; ++v) {
        int b = m0 + v + 8 * half;
        out[base + (size_t)b * DH + col] = hfrag[v];                       // hx
        out[base + (size_t)BATCH * DH + (size_t)b * DH + col] = cfrag[v];  // cx
      }
    }

    // inter-workgroup barrier: monotonic per-step counter (no reset hazard)
    __threadfence();
    __syncthreads();
    if (threadIdx.x == 0) {
      __hip_atomic_fetch_add(&ctr[t], 1u, __ATOMIC_RELEASE, __HIP_MEMORY_SCOPE_AGENT);
      while (__hip_atomic_load(&ctr[t], __ATOMIC_ACQUIRE, __HIP_MEMORY_SCOPE_AGENT) <
             (unsigned)gridDim.x)
        __builtin_amdgcn_s_sleep(1);
    }
    __syncthreads();
    __threadfence();
  }
}

extern "C" void kernel_launch(void* const* d_in, const int* in_sizes, int n_in,
                              void* d_out, int out_size, void* d_ws, size_t ws_size,
                              hipStream_t stream) {
  (void)in_sizes; (void)n_in; (void)out_size; (void)ws_size;
  const float* x   = (const float*)d_in[0];
  const float* Wf  = (const float*)d_in[1];
  const float* bf_ = (const float*)d_in[2];
  const float* Wi  = (const float*)d_in[3];
  const float* bi_ = (const float*)d_in[4];
  const float* Wg  = (const float*)d_in[5];
  const float* bg_ = (const float*)d_in[6];
  const float* Wo  = (const float*)d_in[7];
  const float* bo_ = (const float*)d_in[8];
  float* out = (float*)d_out;

  // workspace carve-out (~170 MB): wbf 4MB | xbf 32MB | zbuf 128MB | hbuf 128KB | ctr 2KB
  char* ws = (char*)d_ws;
  size_t off = 0;
  auto carve = [&](size_t bytes) {
    size_t p = off;
    off = (off + bytes + 255) & ~(size_t)255;
    return p;
  };
  unsigned short* wbf  = (unsigned short*)(ws + carve((size_t)4 * DH * LDW * 2));
  unsigned short* xbf  = (unsigned short*)(ws + carve((size_t)T_STEPS * BATCH * DIN * 2));
  unsigned short* zbuf = (unsigned short*)(ws + carve((size_t)T_STEPS * 4 * BATCH * DH * 2));
  size_t hbuf_off = carve((size_t)2 * BATCH * DH * 2);
  unsigned short* hbuf = (unsigned short*)(ws + hbuf_off);
  unsigned int* ctr = (unsigned int*)(ws + carve((size_t)T_STEPS * 4));

  // zero h staging (h_{-1} = 0) and all barrier counters; hbuf..ctr contiguous
  hipMemsetAsync(ws + hbuf_off, 0, (size_t)2 * BATCH * DH * 2 + (size_t)T_STEPS * 4, stream);

  {
    size_t total = (size_t)4 * DH * LDW;
    lstm_convert_w<<<(int)((total + 255) / 256), 256, 0, stream>>>(Wf, Wi, Wg, Wo, wbf);
  }
  {
    size_t total = (size_t)T_STEPS * BATCH * DIN / 8;   // 8 floats per thread
    lstm_convert_x<<<(int)(total / 256), 256, 0, stream>>>(x, xbf);
  }
  lstm_xgemm<<<dim3((4 * DH) / 128, (T_STEPS * BATCH) / 128), 256, 0, stream>>>(xbf, wbf, zbuf);
  lstm_recurrent<<<NWG2, 256, 0, stream>>>(zbuf, wbf, bf_, bi_, bg_, bo_, hbuf, out, ctr);
}
